// Tacotron2Hybrid_89215060672831
// MI455X (gfx1250) — matrix operand order repaired and checked by reading, not compiled, not run on hardware
//
#include <hip/hip_runtime.h>

// ---------------- model constants ----------------
#define BB   32
#define SS   384
#define TT   1000
#define DEC  512
#define STY  128

// ---------------- WMMA fragment machinery ----------------
typedef __attribute__((ext_vector_type(16))) __bf16        v16bf;
typedef __attribute__((ext_vector_type(8)))  float         v8f;
typedef __attribute__((ext_vector_type(4)))  unsigned int  v4u;
typedef __attribute__((ext_vector_type(8)))  unsigned int  v8u;
typedef __attribute__((ext_vector_type(4)))  int           v4i;

union Frag { v8u u; v16bf bf; };

__device__ __forceinline__ unsigned short f2bf(float f) {
  unsigned int u = __float_as_uint(f);
  unsigned int r = ((u >> 16) & 1u) + 0x7FFFu;   // round-to-nearest-even
  return (unsigned short)((u + r) >> 16);
}
__device__ __forceinline__ float bf2f(unsigned short h) {
  return __uint_as_float(((unsigned int)h) << 16);
}
__device__ __forceinline__ float sigf(float x) { return 1.f / (1.f + __expf(-x)); }
__device__ __forceinline__ float tanhf_(float x) {
  float e = __expf(-2.f * fabsf(x));
  float t = (1.f - e) / (1.f + e);
  return copysignf(t, x);
}

// A fragment (16x32 bf16, M rows striped per lane). ISA 7.12.2.
__device__ __forceinline__ v16bf load_a(const unsigned short* A, long lda, int kbase) {
  int l = threadIdx.x & 31; int m = l & 15; int hf = l >> 4;
  const unsigned short* p = A + (long)m * lda + kbase + hf * 8;
  v4u lo = *(const v4u*)p;
  v4u hi = *(const v4u*)(p + 16);
  Frag f;
  f.u[0]=lo[0]; f.u[1]=lo[1]; f.u[2]=lo[2]; f.u[3]=lo[3];
  f.u[4]=hi[0]; f.u[5]=hi[1]; f.u[6]=hi[2]; f.u[7]=hi[3];
  return f.bf;
}
// B fragment (32x16 bf16) from (N,K) row-major weights.
__device__ __forceinline__ v16bf load_b(const unsigned short* W, long ldb, int kbase) {
  int l = threadIdx.x & 31; int n = l & 15; int hf = l >> 4;
  const unsigned short* p = W + (long)n * ldb + kbase + hf * 8;
  v4u x = *(const v4u*)p;
  v4u y = *(const v4u*)(p + 16);
  Frag f;
  f.u[0]=x[0]; f.u[1]=x[1]; f.u[2]=x[2]; f.u[3]=x[3];
  f.u[4]=y[0]; f.u[5]=y[1]; f.u[6]=y[2]; f.u[7]=y[3];
  return f.bf;
}
__device__ __forceinline__ v8f wmma_bf(v16bf a, v16bf b, v8f c) {
  return __builtin_amdgcn_wmma_f32_16x16x32_bf16(false, a, false, b, (short)0, c, false, false);
}
__device__ __forceinline__ v8f gemm_tile(const unsigned short* A, long lda,
                                         const unsigned short* W, long ldb,
                                         int K, v8f acc) {
  for (int k = 0; k < K; k += 32)
    acc = wmma_bf(load_a(A, lda, k), load_b(W, ldb, k), acc);
  return acc;
}
// 4 output tiles sharing one A fragment (gate fusion / N-blocking):
// tile i uses weight rows at W + i*gstr.
__device__ __forceinline__ void gemm4(const unsigned short* A, long lda,
                                      const unsigned short* W, long ldb, long gstr,
                                      int K, v8f* G) {
  for (int k = 0; k < K; k += 32) {
    v16bf af = load_a(A, lda, k);
#pragma unroll
    for (int i = 0; i < 4; i++)
      G[i] = wmma_bf(af, load_b(W + gstr * i, ldb, k), G[i]);
  }
}

// ---------------- small utility kernels ----------------
__global__ void k_cast(const float* src, unsigned short* dst, long n) {
  long i = (long)blockIdx.x * blockDim.x + threadIdx.x;
  if (i < n) dst[i] = f2bf(src[i]);
}
__global__ void k_castpad(const float* src, unsigned short* dst, long rows, int cin, int cpad) {
  long i = (long)blockIdx.x * blockDim.x + threadIdx.x;
  if (i >= rows * (long)cpad) return;
  long r = i / cpad; int c0 = (int)(i % cpad);
  dst[i] = (c0 < cin) ? f2bf(src[r * cin + c0]) : (unsigned short)0;
}
// conv weight (O,I,T) f32 -> [T][O][Ipad] bf16 (zero padded)
__global__ void k_repack_conv(const float* W, unsigned short* dst, int O, int I, int T, int Ipad) {
  long n = (long)T * O * Ipad;
  long i = (long)blockIdx.x * blockDim.x + threadIdx.x;
  if (i >= n) return;
  int t = (int)(i / ((long)O * Ipad));
  long r = i % ((long)O * Ipad);
  int o = (int)(r / Ipad), ip = (int)(r % Ipad);
  dst[i] = (ip < I) ? f2bf(W[((long)o * I + ip) * T + t]) : (unsigned short)0;
}
__global__ void k_embed(const int* text, const unsigned short* embB, unsigned short* X0, long BS) {
  long i = (long)blockIdx.x * blockDim.x + threadIdx.x;
  if (i >= BS * 512) return;
  long row = i >> 9; int c = (int)(i & 511);
  X0[i] = embB[(long)text[row] * 512 + c];
}
__global__ void k_zero(unsigned int* p, long n) {
  long i = (long)blockIdx.x * blockDim.x + threadIdx.x;
  if (i < n) p[i] = 0u;
}
__global__ void k_hcat(const unsigned short* Hs, unsigned short* Hcat, int H, int Slen) {
  int i = blockIdx.x * blockDim.x + threadIdx.x;
  if (i >= 32 * 2 * H) return;
  int b = i / (2 * H), col = i % (2 * H);
  int s = (col < H) ? (Slen - 1) : 0;
  Hcat[(long)b * 2 * H + col] = Hs[((long)b * Slen + s) * (2 * H) + col];
}
__global__ void k_durlin(const unsigned short* D, const float* W, const float* b, float* out, int n) {
  int i = blockIdx.x * blockDim.x + threadIdx.x;
  if (i >= n) return;
  float acc = b[0];
  for (int k = 0; k < 256; k++) acc += bf2f(D[(long)i * 256 + k]) * W[k];
  out[i] = fmaxf(acc, 0.f);
}

// ---------------- generic GEMM: each wave computes 16x64 (4 N-tiles, A reuse) ----------------
__global__ __launch_bounds__(256) void k_gemm(const unsigned short* A, long lda,
                                              const unsigned short* W, long ldb,
                                              const float* bias, int relu,
                                              unsigned short* outB, float* outF, long ldc,
                                              int Mt, int Nt4, int K) {
  int t = blockIdx.x * (blockDim.x >> 5) + (threadIdx.x >> 5);
  if (t >= Mt * Nt4) return;
  int mt = t / Nt4, n4 = t % Nt4;
  const unsigned short* Ab = A + (long)mt * 16 * lda;
  const unsigned short* Wb = W + (long)n4 * 64 * ldb;
  v8f z = {0.f,0.f,0.f,0.f,0.f,0.f,0.f,0.f};
  v8f G[4] = {z, z, z, z};
  gemm4(Ab, lda, Wb, ldb, (long)16 * ldb, K, G);
  int l = threadIdx.x & 31, n = l & 15, mh = l >> 4;
#pragma unroll
  for (int i = 0; i < 4; i++) {
    int nc = n4 * 64 + i * 16 + n;
    float bv = bias ? bias[nc] : 0.f;
#pragma unroll
    for (int j = 0; j < 8; j++) {
      float v = G[i][j] + bv;
      if (relu) v = fmaxf(v, 0.f);
      long row = (long)mt * 16 + mh * 8 + j;
      long idx = row * ldc + nc;
      if (outB) outB[idx] = f2bf(v);
      if (outF) outF[idx] = v;
    }
  }
}

// ---------------- conv1d as per-tap shifted WMMA GEMM (always ReLU) ----------------
__device__ __forceinline__ v16bf load_a_shift(const unsigned short* Xb, long Cin, int kbase,
                                              int sbase, int shift, int S) {
  int l = threadIdx.x & 31; int m = l & 15; int hf = l >> 4;
  int s = sbase + m + shift;
  Frag f;
  if ((unsigned)s < (unsigned)S) {
    const unsigned short* p = Xb + (long)s * Cin + kbase + hf * 8;
    v4u lo = *(const v4u*)p;
    v4u hi = *(const v4u*)(p + 16);
    f.u[0]=lo[0]; f.u[1]=lo[1]; f.u[2]=lo[2]; f.u[3]=lo[3];
    f.u[4]=hi[0]; f.u[5]=hi[1]; f.u[6]=hi[2]; f.u[7]=hi[3];
  } else {
#pragma unroll
    for (int i = 0; i < 8; i++) f.u[i] = 0u;
  }
  return f.bf;
}
__global__ __launch_bounds__(256) void k_conv(const unsigned short* X, int Cin,
                                              const unsigned short* Wt, const float* bias,
                                              unsigned short* Y, int Cout,
                                              int S, int taps, int pad, int Mt) {
  int Nt4 = Cout >> 6;                       // 4 N-tiles per wave
  int t = blockIdx.x * (blockDim.x >> 5) + (threadIdx.x >> 5);
  if (t >= Mt * Nt4) return;
  int mt = t / Nt4, n4 = t % Nt4;
  int b = (mt * 16) / S, sbase = (mt * 16) % S;
  const unsigned short* Xb = X + (long)b * S * Cin;
  v8f z = {0.f,0.f,0.f,0.f,0.f,0.f,0.f,0.f};
  v8f G[4] = {z, z, z, z};
  for (int tap = 0; tap < taps; ++tap) {
    const unsigned short* Wb = Wt + ((long)tap * Cout + (long)n4 * 64) * Cin;
    for (int k = 0; k < Cin; k += 32) {
      v16bf af = load_a_shift(Xb, Cin, k, sbase, tap - pad, S);
#pragma unroll
      for (int i = 0; i < 4; i++)
        G[i] = wmma_bf(af, load_b(Wb + (long)i * 16 * Cin, Cin, k), G[i]);
    }
  }
  int l = threadIdx.x & 31, n = l & 15, mh = l >> 4;
#pragma unroll
  for (int i = 0; i < 4; i++) {
    int nc = n4 * 64 + i * 16 + n;
    float bv = bias[nc];
#pragma unroll
    for (int j = 0; j < 8; j++) {
      float v = fmaxf(G[i][j] + bv, 0.f);
      long row = (long)b * S + sbase + mh * 8 + j;
      Y[row * Cout + nc] = f2bf(v);
    }
  }
}

// ---------------- bidirectional LSTM recurrence (Gx precomputed) ----------------
// blockIdx.x = dir. h kept in LDS; c kept in VGPRs; 4 gates fused on one A fragment.
__global__ __launch_bounds__(1024) void k_lstm(const float* GxF, const float* GxB,
                                               const unsigned short* WhhF, const unsigned short* WhhB,
                                               const float* bhhF, const float* bhhB,
                                               unsigned short* Hout, int H, int Slen) {
  __shared__ unsigned short h_lds[32 * 256];
  int dir = blockIdx.x;
  const float* Gx = dir ? GxB : GxF;
  const unsigned short* Whh = dir ? WhhB : WhhF;
  const float* bhh = dir ? bhhB : bhhF;
  int wid = threadIdx.x >> 5;
  int mTile = wid & 1;
  int hTile = wid >> 1;
  int l = threadIdx.x & 31, n = l & 15, mh = l >> 4;
  for (int i = threadIdx.x; i < 32 * H; i += blockDim.x) h_lds[i] = 0;
  float c[8];
#pragma unroll
  for (int j = 0; j < 8; j++) c[j] = 0.f;
  int H4 = 4 * H;
  const unsigned short* Ab = &h_lds[mTile * 16 * H];
  const unsigned short* Wbase = Whh + (long)(hTile * 16) * H;   // gate gi at + gi*H*H
  for (int it = 0; it < Slen; ++it) {
    int s = dir ? (Slen - 1 - it) : it;
    __syncthreads();                           // h_lds stable
    v8f z = {0.f,0.f,0.f,0.f,0.f,0.f,0.f,0.f};
    v8f G[4] = {z, z, z, z};
    gemm4(Ab, H, Wbase, H, (long)H * H, H, G);
#pragma unroll
    for (int gi = 0; gi < 4; ++gi) {
      int nc = gi * H + hTile * 16;
      float bb = bhh[nc + n];
#pragma unroll
      for (int j = 0; j < 8; j++) {
        int b = mTile * 16 + mh * 8 + j;
        G[gi][j] += Gx[((long)b * Slen + s) * H4 + nc + n] + bb;
      }
    }
    __syncthreads();                           // all h reads done
#pragma unroll
    for (int j = 0; j < 8; j++) {
      float ig = sigf(G[0][j]), fg = sigf(G[1][j]);
      float gg = tanhf_(G[2][j]), og = sigf(G[3][j]);
      c[j] = fg * c[j] + ig * gg;
      float h = og * tanhf_(c[j]);
      int b = mTile * 16 + mh * 8 + j;
      int col = hTile * 16 + n;
      unsigned short hb = f2bf(h);
      h_lds[b * H + col] = hb;
      Hout[((long)b * Slen + s) * (2 * H) + (long)dir * H + col] = hb;
    }
  }
}

// ---------------- persistent autoregressive decoder ----------------
struct DecArgs {
  const unsigned short *Kb, *Vb, *P, *style;
  const unsigned short *Wq, *Wo, *Wih1, *Whh1, *Wih2, *Whh2, *Wmel;
  const float *bq, *bo, *bih1, *bhh1, *bih2, *bhh2, *bmel;
  const float *wgate, *bgate;
  unsigned short *qg, *ctx0g, *ctxg;
  unsigned short *h1a, *h1b, *h2a, *h2b;
  float *omel, *ogate;
  unsigned int* cnt;
};

__device__ void gbar(unsigned int* cnt, unsigned int nwg, unsigned int* phase) {
  __syncthreads();
  if (threadIdx.x == 0) {
    __threadfence();
    unsigned int ph = ++(*phase);
    atomicAdd(cnt, 1u);
    while (atomicAdd(cnt, 0u) < ph * nwg) __builtin_amdgcn_s_sleep(1);
  }
  __syncthreads();
  __threadfence();
}

#if defined(__has_builtin)
#if __has_builtin(__builtin_amdgcn_global_load_async_to_lds_b128)
#define HAVE_ASYNC_LDS 1
#endif
#endif

__global__ __launch_bounds__(128) void k_decoder(DecArgs a) {
  __shared__ float qs[DEC];
  __shared__ float sc[SS];
  __shared__ float red[128];
  __shared__ unsigned short styS[32 * STY];   // style operand resident in LDS
  __shared__ unsigned int phase;
  if (threadIdx.x == 0) phase = 0u;

  // One-time stage of the style matrix (read by every step) via CDNA5 async path.
#ifdef HAVE_ASYNC_LDS
  for (int i = threadIdx.x * 8; i < 32 * STY; i += 128 * 8)
    __builtin_amdgcn_global_load_async_to_lds_b128(
        (v4i*)(unsigned short*)(a.style + i), (v4i*)&styS[i], 0, 0);
#if __has_builtin(__builtin_amdgcn_s_wait_asynccnt)
  __builtin_amdgcn_s_wait_asynccnt(0);
#else
  asm volatile("s_wait_asynccnt 0" ::: "memory");
#endif
#else
  for (int i = threadIdx.x; i < 32 * STY; i += 128) styS[i] = a.style[i];
#endif
  __syncthreads();

  const int wid = threadIdx.x >> 5;
  const int gw = blockIdx.x * 4 + wid;
  const int l = threadIdx.x & 31;
  const int ln = l & 15, lh = l >> 4;
  float c1[8], c2[8];
#pragma unroll
  for (int j = 0; j < 8; j++) { c1[j] = 0.f; c2[j] = 0.f; }
  const float scl = 0.0883883476483184f;  // 1/sqrt(128)
  const v8f z = {0.f,0.f,0.f,0.f,0.f,0.f,0.f,0.f};

  for (int t = 0; t < TT; ++t) {
    const unsigned short* h1o = (t & 1) ? a.h1b : a.h1a;
    unsigned short*       h1n = (t & 1) ? a.h1a : a.h1b;
    const unsigned short* h2o = (t & 1) ? a.h2b : a.h2a;
    unsigned short*       h2n = (t & 1) ? a.h2a : a.h2b;

    // ---- P1: q = h2 @ Wq^T + bq  (2 x 8 wave-tiles of 16x64) ----
    if (gw < 16) {
      int mt = gw >> 3, n4 = gw & 7;
      v8f G[4] = {z, z, z, z};
      gemm4(h2o + mt * 16 * DEC, DEC, a.Wq + (long)n4 * 64 * DEC, DEC, (long)16 * DEC, DEC, G);
#pragma unroll
      for (int i = 0; i < 4; i++) {
        int nc = n4 * 64 + i * 16 + ln;
        float bv = a.bq[nc];
#pragma unroll
        for (int j = 0; j < 8; j++) {
          int b = mt * 16 + lh * 8 + j;
          a.qg[b * DEC + nc] = f2bf(G[i][j] + bv);
        }
      }
    }
    gbar(a.cnt, 64u, &phase);

    // ---- P2: scaled-dot attention, one workgroup per batch row ----
    if (blockIdx.x < 32) {
      int b = blockIdx.x;
      for (int d0 = threadIdx.x; d0 < DEC; d0 += 128) qs[d0] = bf2f(a.qg[b * DEC + d0]);
      __syncthreads();
      for (int s0 = threadIdx.x; s0 < SS; s0 += 128) {
        const unsigned short* kp = a.Kb + ((long)b * SS + s0) * DEC;
        __builtin_prefetch(kp + DEC, 0, 1);
        float d = 0.f;
        for (int k2 = 0; k2 < DEC; k2++) d += bf2f(kp[k2]) * qs[k2];
        sc[s0] = d * scl;
      }
      __syncthreads();
      float mx = -1e30f;
      for (int s0 = threadIdx.x; s0 < SS; s0 += 128) mx = fmaxf(mx, sc[s0]);
      red[threadIdx.x] = mx;
      __syncthreads();
      for (int off = 64; off > 0; off >>= 1) {
        if (threadIdx.x < off) red[threadIdx.x] = fmaxf(red[threadIdx.x], red[threadIdx.x + off]);
        __syncthreads();
      }
      mx = red[0];
      __syncthreads();
      float sum = 0.f;
      for (int s0 = threadIdx.x; s0 < SS; s0 += 128) {
        float e = __expf(sc[s0] - mx); sc[s0] = e; sum += e;
      }
      red[threadIdx.x] = sum;
      __syncthreads();
      for (int off = 64; off > 0; off >>= 1) {
        if (threadIdx.x < off) red[threadIdx.x] += red[threadIdx.x + off];
        __syncthreads();
      }
      float inv = 1.f / red[0];
      __syncthreads();
      for (int d0 = threadIdx.x; d0 < DEC; d0 += 128) {
        const unsigned short* vp = a.Vb + (long)b * SS * DEC + d0;
        float av = 0.f;
        for (int s0 = 0; s0 < SS; s0++) av += sc[s0] * bf2f(vp[(long)s0 * DEC]);
        a.ctx0g[b * DEC + d0] = f2bf(av * inv);
      }
    }
    gbar(a.cnt, 64u, &phase);

    // ---- P3: ctx = ctx0 @ Wo^T + bo ----
    if (gw < 16) {
      int mt = gw >> 3, n4 = gw & 7;
      v8f G[4] = {z, z, z, z};
      gemm4(a.ctx0g + mt * 16 * DEC, DEC, a.Wo + (long)n4 * 64 * DEC, DEC, (long)16 * DEC, DEC, G);
#pragma unroll
      for (int i = 0; i < 4; i++) {
        int nc = n4 * 64 + i * 16 + ln;
        float bv = a.bo[nc];
#pragma unroll
        for (int j = 0; j < 8; j++) {
          int b = mt * 16 + lh * 8 + j;
          a.ctxg[b * DEC + nc] = f2bf(G[i][j] + bv);
        }
      }
    }
    gbar(a.cnt, 64u, &phase);

    // ---- P4: LSTM1, inp = [prenet(t) | ctx | style], 4 gates fused ----
    if (gw < 64) {
      int mt = gw & 1, ht = gw >> 1;
      v8f G[4] = {z, z, z, z};
      const unsigned short* W1 = a.Wih1 + (long)(ht * 16) * 896;
      gemm4(a.P + (long)t * 256 + (long)mt * 16 * ((long)TT * 256), (long)TT * 256,
            W1, 896, (long)DEC * 896, 256, G);
      gemm4(a.ctxg + mt * 16 * DEC, DEC, W1 + 256, 896, (long)DEC * 896, DEC, G);
      gemm4(styS + mt * 16 * STY, STY, W1 + 768, 896, (long)DEC * 896, STY, G);
      gemm4(h1o + mt * 16 * DEC, DEC, a.Whh1 + (long)(ht * 16) * DEC, DEC, (long)DEC * DEC, DEC, G);
#pragma unroll
      for (int gi = 0; gi < 4; ++gi) {
        int nc = gi * DEC + ht * 16 + ln;
        float bb = a.bih1[nc] + a.bhh1[nc];
#pragma unroll
        for (int j = 0; j < 8; j++) G[gi][j] += bb;
      }
#pragma unroll
      for (int j = 0; j < 8; j++) {
        float ig = sigf(G[0][j]), fg = sigf(G[1][j]);
        float gg = tanhf_(G[2][j]), og = sigf(G[3][j]);
        c1[j] = fg * c1[j] + ig * gg;
        float hh = og * tanhf_(c1[j]);
        int b = mt * 16 + lh * 8 + j;
        h1n[b * DEC + ht * 16 + ln] = f2bf(hh);
      }
    }
    gbar(a.cnt, 64u, &phase);

    // ---- P5: LSTM2, 4 gates fused ----
    if (gw < 64) {
      int mt = gw & 1, ht = gw >> 1;
      v8f G[4] = {z, z, z, z};
      gemm4(h1n + mt * 16 * DEC, DEC, a.Wih2 + (long)(ht * 16) * DEC, DEC, (long)DEC * DEC, DEC, G);
      gemm4(h2o + mt * 16 * DEC, DEC, a.Whh2 + (long)(ht * 16) * DEC, DEC, (long)DEC * DEC, DEC, G);
#pragma unroll
      for (int gi = 0; gi < 4; ++gi) {
        int nc = gi * DEC + ht * 16 + ln;
        float bb = a.bih2[nc] + a.bhh2[nc];
#pragma unroll
        for (int j = 0; j < 8; j++) G[gi][j] += bb;
      }
#pragma unroll
      for (int j = 0; j < 8; j++) {
        float ig = sigf(G[0][j]), fg = sigf(G[1][j]);
        float gg = tanhf_(G[2][j]), og = sigf(G[3][j]);
        c2[j] = fg * c2[j] + ig * gg;
        float hh = og * tanhf_(c2[j]);
        int b = mt * 16 + lh * 8 + j;
        h2n[b * DEC + ht * 16 + ln] = f2bf(hh);
      }
    }
    gbar(a.cnt, 64u, &phase);

    // ---- P6: mel = [h2|style] @ Wmel^T + bmel ; gate = h2 . wg + bg ----
    if (gw < 10) {
      int mt = gw & 1, nt = gw >> 1;   // nt in 0..4, 5*16 = 80 mel bins
      int nc = nt * 16;
      v8f acc = z;
      acc = gemm_tile(h2n + mt * 16 * DEC, DEC, a.Wmel + (long)nc * 640, 640, DEC, acc);
      acc = gemm_tile(styS + mt * 16 * STY, STY, a.Wmel + (long)nc * 640 + 512, 640, STY, acc);
      float bv = a.bmel[nc + ln];
#pragma unroll
      for (int j = 0; j < 8; j++) {
        int b = mt * 16 + lh * 8 + j;
        a.omel[((long)b * TT + t) * 80 + nc + ln] = acc[j] + bv;
      }
    } else if (gw == 10) {
      int b = l;
      float accv = a.bgate[0];
      const unsigned short* hp = h2n + b * DEC;
      for (int k2 = 0; k2 < DEC; k2++) accv += bf2f(hp[k2]) * a.wgate[k2];
      a.ogate[(long)b * TT + t] = accv;
    }
    gbar(a.cnt, 64u, &phase);
  }
}

// ==================================================================
extern "C" void kernel_launch(void* const* d_in, const int* in_sizes, int n_in,
                              void* d_out, int out_size, void* d_ws, size_t ws_size,
                              hipStream_t stream) {
  (void)in_sizes; (void)n_in; (void)out_size; (void)ws_size;
#define PF(i) ((const float*)d_in[i])
  const int* text = (const int*)d_in[0];
  const float* mels = PF(2);

  const long BS = (long)BB * SS;   // 12288
  const long BT = (long)BB * TT;   // 32000

  // ---- workspace bump allocator ----
  size_t off = 0;
  auto wsa = [&](size_t bytes) -> void* {
    off = (off + 255) & ~(size_t)255;
    void* p = (char*)d_ws + off;
    off += bytes;
    return p;
  };
  typedef unsigned short u16;
  // weights (bf16)
  u16* embB  = (u16*)wsa(2L * 132 * 512);
  u16* encW1 = (u16*)wsa(2L * 5 * 512 * 512);
  u16* encW2 = (u16*)wsa(2L * 5 * 512 * 512);
  u16* encW3 = (u16*)wsa(2L * 5 * 512 * 512);
  u16 *WihE[4], *WhhE[4];
  for (int i = 0; i < 4; i++) { WihE[i] = (u16*)wsa(2L * 1024 * 512); WhhE[i] = (u16*)wsa(2L * 1024 * 256); }
  u16* styW1 = (u16*)wsa(2L * 3 * 256 * 96);
  u16* styW2 = (u16*)wsa(2L * 3 * 256 * 256);
  u16 *WihS[4], *WhhS[4];
  for (int i = 0; i < 4; i++) { WihS[i] = (u16*)wsa(2L * 512 * 256); WhhS[i] = (u16*)wsa(2L * 512 * 128); }
  u16* spW   = (u16*)wsa(2L * 128 * 256);
  u16* durW1 = (u16*)wsa(2L * 3 * 256 * 512);
  u16* durW2 = (u16*)wsa(2L * 3 * 256 * 256);
  u16* pre1W = (u16*)wsa(2L * 256 * 96);
  u16* pre2W = (u16*)wsa(2L * 256 * 256);
  u16* dWih1 = (u16*)wsa(2L * 2048 * 896);
  u16* dWhh1 = (u16*)wsa(2L * 2048 * 512);
  u16* dWih2 = (u16*)wsa(2L * 2048 * 512);
  u16* dWhh2 = (u16*)wsa(2L * 2048 * 512);
  u16* WqB = (u16*)wsa(2L * 512 * 512);
  u16* WkB = (u16*)wsa(2L * 512 * 512);
  u16* WvB = (u16*)wsa(2L * 512 * 512);
  u16* WoB = (u16*)wsa(2L * 512 * 512);
  u16* WmelB = (u16*)wsa(2L * 80 * 640);
  // activations
  u16* X0 = (u16*)wsa(2L * BS * 512);
  u16* Xa = (u16*)wsa(2L * BS * 512);
  u16* Xb = (u16*)wsa(2L * BS * 512);
  u16* Ya = (u16*)wsa(2L * BT * 256);
  u16* Yb = (u16*)wsa(2L * BT * 256);
  u16* Pbuf = (u16*)wsa(2L * BT * 256);
  u16* Kb = (u16*)wsa(2L * BS * 512);
  u16* Vb = (u16*)wsa(2L * BS * 512);
  float* GxF = (float*)wsa(4L * BT * 512);   // >= enc's 12288*1024 too
  float* GxB = (float*)wsa(4L * BT * 512);
  u16* Hcat = (u16*)wsa(2L * 32 * 256);
  u16* styleB = (u16*)wsa(2L * 32 * 128);
  u16* qg = (u16*)wsa(2L * 32 * 512);
  u16* ctx0g = (u16*)wsa(2L * 32 * 512);
  u16* ctxg = (u16*)wsa(2L * 32 * 512);
  unsigned char* zone = (unsigned char*)wsa(256 + 4L * 32 * 512 * 2);
  unsigned int* cnt = (unsigned int*)zone;
  u16* h1a = (u16*)(zone + 256);
  u16* h1b = (u16*)(zone + 256 + 1 * 32768);
  u16* h2a = (u16*)(zone + 256 + 2 * 32768);
  u16* h2b = (u16*)(zone + 256 + 3 * 32768);

  auto cast = [&](int idx, u16* dst, long n) {
    k_cast<<<(unsigned)((n + 255) / 256), 256, 0, stream>>>(PF(idx), dst, n);
  };
  auto gemm = [&](const u16* A, long lda, const u16* W, long ldb, const float* bias, int relu,
                  u16* outB, float* outF, long ldc, int Mt, int Nt, int K) {
    int tiles = Mt * (Nt >> 2);              // waves of 16x64
    k_gemm<<<(tiles + 7) / 8, 256, 0, stream>>>(A, lda, W, ldb, bias, relu, outB, outF, ldc,
                                                Mt, Nt >> 2, K);
  };
  auto conv = [&](const u16* X, int Cin, const u16* Wt, const float* bias, u16* Y,
                  int Cout, int S, int taps, int pad, int Mt) {
    int tiles = Mt * (Cout >> 6);
    k_conv<<<(tiles + 7) / 8, 256, 0, stream>>>(X, Cin, Wt, bias, Y, Cout, S, taps, pad, Mt);
  };

  // ---- weight preparation (bf16 cast / per-tap repack) ----
  cast(3, embB, 132L * 512);
  k_repack_conv<<<(unsigned)((5L*512*512 + 255)/256),256,0,stream>>>(PF(4), encW1, 512, 512, 5, 512);
  k_repack_conv<<<(unsigned)((5L*512*512 + 255)/256),256,0,stream>>>(PF(6), encW2, 512, 512, 5, 512);
  k_repack_conv<<<(unsigned)((5L*512*512 + 255)/256),256,0,stream>>>(PF(8), encW3, 512, 512, 5, 512);
  const int encIdx[4] = {10, 14, 18, 22};   // l0f, l0b, l1f, l1b
  for (int i = 0; i < 4; i++) { cast(encIdx[i], WihE[i], 1024L*512); cast(encIdx[i]+1, WhhE[i], 1024L*256); }
  k_repack_conv<<<(unsigned)((3L*256*96 + 255)/256),256,0,stream>>>(PF(26), styW1, 256, 80, 3, 96);
  k_repack_conv<<<(unsigned)((3L*256*256 + 255)/256),256,0,stream>>>(PF(28), styW2, 256, 256, 3, 256);
  const int styIdx[4] = {30, 34, 38, 42};
  for (int i = 0; i < 4; i++) { cast(styIdx[i], WihS[i], 512L*256); cast(styIdx[i]+1, WhhS[i], 512L*128); }
  cast(46, spW, 128L * 256);
  k_repack_conv<<<(unsigned)((3L*256*512 + 255)/256),256,0,stream>>>(PF(48), durW1, 256, 512, 3, 512);
  k_repack_conv<<<(unsigned)((3L*256*256 + 255)/256),256,0,stream>>>(PF(50), durW2, 256, 256, 3, 256);
  k_castpad<<<(unsigned)((256L*96 + 255)/256),256,0,stream>>>(PF(54), pre1W, 256, 80, 96);
  cast(56, pre2W, 256L * 256);
  cast(58, dWih1, 2048L * 896); cast(59, dWhh1, 2048L * 512);
  cast(62, dWih2, 2048L * 512); cast(63, dWhh2, 2048L * 512);
  cast(66, WqB, 512L * 512); cast(68, WkB, 512L * 512);
  cast(70, WvB, 512L * 512); cast(72, WoB, 512L * 512);
  cast(74, WmelB, 80L * 640);

  // ---- encoder ----
  k_embed<<<(unsigned)((BS * 512 + 255) / 256), 256, 0, stream>>>(text, embB, X0, BS);
  conv(X0, 512, encW1, PF(5), Xa, 512, SS, 5, 2, (int)(BS / 16));
  conv(Xa, 512, encW2, PF(7), Xb, 512, SS, 5, 2, (int)(BS / 16));
  conv(Xb, 512, encW3, PF(9), X0, 512, SS, 5, 2, (int)(BS / 16));
  // biLSTM layer 0
  gemm(X0, 512, WihE[0], 512, PF(12), 0, nullptr, GxF, 1024, (int)(BS/16), 64, 512);
  gemm(X0, 512, WihE[1], 512, PF(16), 0, nullptr, GxB, 1024, (int)(BS/16), 64, 512);
  k_lstm<<<2, 1024, 0, stream>>>(GxF, GxB, WhhE[0], WhhE[1], PF(13), PF(17), Xa, 256, SS);
  // biLSTM layer 1 -> enc_out in Xb
  gemm(Xa, 512, WihE[2], 512, PF(20), 0, nullptr, GxF, 1024, (int)(BS/16), 64, 512);
  gemm(Xa, 512, WihE[3], 512, PF(24), 0, nullptr, GxB, 1024, (int)(BS/16), 64, 512);
  k_lstm<<<2, 1024, 0, stream>>>(GxF, GxB, WhhE[2], WhhE[3], PF(21), PF(25), Xb, 256, SS);

  // ---- duration predictor ----
  conv(Xb, 512, durW1, PF(49), Xa, 256, SS, 3, 1, (int)(BS / 16));
  conv(Xa, 256, durW2, PF(51), X0, 256, SS, 3, 1, (int)(BS / 16));
  k_durlin<<<(unsigned)((BS + 255) / 256), 256, 0, stream>>>(X0, PF(52), PF(53),
                                                             (float*)d_out + 2592000, (int)BS);

  // ---- attention K/V ----
  gemm(Xb, 512, WkB, 512, PF(69), 0, Kb, nullptr, 512, (int)(BS/16), 32, 512);
  gemm(Xb, 512, WvB, 512, PF(71), 0, Vb, nullptr, 512, (int)(BS/16), 32, 512);

  // ---- style encoder ----
  k_castpad<<<(unsigned)((BT * 96 + 255) / 256), 256, 0, stream>>>(mels, Xa, BT, 80, 96);
  conv(Xa, 96, styW1, PF(27), Ya, 256, TT, 3, 1, (int)(BT / 16));
  conv(Ya, 256, styW2, PF(29), Yb, 256, TT, 3, 1, (int)(BT / 16));
  gemm(Yb, 256, WihS[0], 256, PF(32), 0, nullptr, GxF, 512, (int)(BT/16), 32, 256);
  gemm(Yb, 256, WihS[1], 256, PF(36), 0, nullptr, GxB, 512, (int)(BT/16), 32, 256);
  k_lstm<<<2, 512, 0, stream>>>(GxF, GxB, WhhS[0], WhhS[1], PF(33), PF(37), Ya, 128, TT);
  gemm(Ya, 256, WihS[2], 256, PF(40), 0, nullptr, GxF, 512, (int)(BT/16), 32, 256);
  gemm(Ya, 256, WihS[3], 256, PF(44), 0, nullptr, GxB, 512, (int)(BT/16), 32, 256);
  k_lstm<<<2, 512, 0, stream>>>(GxF, GxB, WhhS[2], WhhS[3], PF(41), PF(45), Yb, 128, TT);
  k_hcat<<<32, 256, 0, stream>>>(Yb, Hcat, 128, TT);
  gemm(Hcat, 256, spW, 256, PF(47), 0, styleB, (float*)d_out + 2604288, 128, 2, 8, 256);

  // ---- prenet for all timesteps (teacher forcing) ----
  gemm(Xa, 96, pre1W, 96, PF(55), 1, Ya, nullptr, 256, (int)(BT/16), 16, 96);
  gemm(Ya, 256, pre2W, 256, PF(57), 1, Pbuf, nullptr, 256, (int)(BT/16), 16, 256);

  // ---- persistent decoder ----
  k_zero<<<(32832 + 255) / 256, 256, 0, stream>>>((unsigned int*)zone, 32832);
  DecArgs da;
  da.Kb = Kb; da.Vb = Vb; da.P = Pbuf; da.style = styleB;
  da.Wq = WqB; da.Wo = WoB; da.Wih1 = dWih1; da.Whh1 = dWhh1;
  da.Wih2 = dWih2; da.Whh2 = dWhh2; da.Wmel = WmelB;
  da.bq = PF(67); da.bo = PF(73);
  da.bih1 = PF(60); da.bhh1 = PF(61); da.bih2 = PF(64); da.bhh2 = PF(65);
  da.bmel = PF(75); da.wgate = PF(76); da.bgate = PF(77);
  da.qg = qg; da.ctx0g = ctx0g; da.ctxg = ctxg;
  da.h1a = h1a; da.h1b = h1b; da.h2a = h2a; da.h2b = h2b;
  da.omel = (float*)d_out; da.ogate = (float*)d_out + 2560000;
  da.cnt = cnt;
  k_decoder<<<64, 128, 0, stream>>>(da);
#undef PF
}
